// MHA_58884001628577
// MI455X (gfx1250) — matrix operand order repaired and checked by reading, not compiled, not run on hardware
//
#include <hip/hip_runtime.h>
#include <hip/hip_bf16.h>
#include <math.h>

#ifndef __has_builtin
#define __has_builtin(x) 0
#endif

typedef __attribute__((ext_vector_type(16))) __bf16        v16bf;
typedef __attribute__((ext_vector_type(2)))  __bf16        v2bf;
typedef __attribute__((ext_vector_type(8)))  float          v8f;
typedef __attribute__((ext_vector_type(4)))  unsigned int   u32x4;
typedef __attribute__((ext_vector_type(2)))  unsigned int   u32x2;
typedef __attribute__((ext_vector_type(4)))  float          f32x4;
typedef __attribute__((ext_vector_type(4)))  unsigned short u16x4;
typedef __attribute__((__vector_size__(16))) int            v4i;
typedef unsigned short u16;

union Frag16 { v16bf v; u32x4 q[2]; };

#if __has_builtin(__builtin_amdgcn_global_load_async_to_lds_b128)
#define HAVE_ASYNC_LDS 1
#endif

__device__ __forceinline__ void async_copy16(const void* g, void* l) {
#ifdef HAVE_ASYNC_LDS
  __builtin_amdgcn_global_load_async_to_lds_b128(
      (__attribute__((address_space(1))) v4i*)g,
      (__attribute__((address_space(3))) v4i*)l, 0, 0);
#else
  (void)g; (void)l;
#endif
}

// native bf16 conversion (backend selects v_cvt_*_bf16_f32, RNE)
__device__ __forceinline__ u16 f2bf(float f) {
  return __builtin_bit_cast(u16, (__bf16)f);
}
__device__ __forceinline__ unsigned pack_bf16x2(float lo, float hi) {
  v2bf p; p[0] = (__bf16)lo; p[1] = (__bf16)hi;
  return __builtin_bit_cast(unsigned, p);
}

__device__ __forceinline__ v8f wmma_bf16(v16bf a, v16bf b, v8f c) {
  return __builtin_amdgcn_wmma_f32_16x16x32_bf16(false, a, false, b, (short)0, c, false, false);
}

// ---------------------------------------------------------------------------
// Kernel 1: qkv = x @ W_qkv + b_qkv ; scatter to Q (pre-scaled by 1/8), K, V
// as bf16 in [B*H][S][64] layout.
// Block tile 128(M) x 128(N), K-chunk 32, software-pipelined staging.
// ---------------------------------------------------------------------------
__global__ __launch_bounds__(256) void qkv_gemm_kernel(
    const float* __restrict__ x, const float* __restrict__ W,
    const float* __restrict__ bias,
    u16* __restrict__ Qb, u16* __restrict__ Kb, u16* __restrict__ Vb)
{
  __shared__ u16 As[128][40];   // [m][k], pad to 40 (80B rows, 16B aligned)
  __shared__ u16 Bt[128][40];   // [n][k] transposed

  const int tid = threadIdx.x;
  const int wid = tid >> 5, lane = tid & 31, l15 = lane & 15, h = lane >> 4;
  const int wRow = (wid >> 1) * 32, wCol = (wid & 1) * 64;
  const int M0 = blockIdx.x * 128, N0 = blockIdx.y * 128;

  const v8f vzero = {0.f,0.f,0.f,0.f,0.f,0.f,0.f,0.f};
  v8f acc[2][4];
#pragma unroll
  for (int ti = 0; ti < 2; ++ti)
#pragma unroll
    for (int tj = 0; tj < 4; ++tj) acc[ti][tj] = vzero;

  f32x4 aPre[4], bPre[4];
#pragma unroll
  for (int i = 0; i < 4; ++i) {
    int idx = tid + i * 256;
    int row = idx >> 3, kk4 = (idx & 7) * 4;               // A: 128x32 f32
    aPre[i] = *(const f32x4*)&x[(size_t)(M0 + row) * 1024 + kk4];
    int kk = idx >> 5, nn4 = (idx & 31) * 4;               // B: 32x128 f32
    bPre[i] = *(const f32x4*)&W[(size_t)kk * 3072 + N0 + nn4];
  }

  for (int k0 = 0; k0 < 1024; k0 += 32) {
    // commit prefetched chunk to LDS (bf16, packed pair conversions)
#pragma unroll
    for (int i = 0; i < 4; ++i) {
      int idx = tid + i * 256;
      int row = idx >> 3, kk4 = (idx & 7) * 4;
      u32x2 a2; a2[0] = pack_bf16x2(aPre[i][0], aPre[i][1]);
                a2[1] = pack_bf16x2(aPre[i][2], aPre[i][3]);
      *(u32x2*)&As[row][kk4] = a2;
      int kk = idx >> 5, nn4 = (idx & 31) * 4;
      Bt[nn4+0][kk]=f2bf(bPre[i][0]); Bt[nn4+1][kk]=f2bf(bPre[i][1]);
      Bt[nn4+2][kk]=f2bf(bPre[i][2]); Bt[nn4+3][kk]=f2bf(bPre[i][3]);
    }
    __syncthreads();

    if (k0 + 32 < 1024) {       // prefetch next chunk while WMMAs run
      int kn = k0 + 32;
#pragma unroll
      for (int i = 0; i < 4; ++i) {
        int idx = tid + i * 256;
        int row = idx >> 3, kk4 = (idx & 7) * 4;
        aPre[i] = *(const f32x4*)&x[(size_t)(M0 + row) * 1024 + kn + kk4];
        int kk = idx >> 5, nn4 = (idx & 31) * 4;
        bPre[i] = *(const f32x4*)&W[(size_t)(kn + kk) * 3072 + N0 + nn4];
      }
    }

    Frag16 af[2], bf[4];
#pragma unroll
    for (int t = 0; t < 2; ++t) {
      int row = wRow + t * 16 + l15;
      af[t].q[0] = *(const u32x4*)&As[row][8 * h];
      af[t].q[1] = *(const u32x4*)&As[row][16 + 8 * h];
    }
#pragma unroll
    for (int t = 0; t < 4; ++t) {
      int col = wCol + t * 16 + l15;
      bf[t].q[0] = *(const u32x4*)&Bt[col][8 * h];
      bf[t].q[1] = *(const u32x4*)&Bt[col][16 + 8 * h];
    }
#pragma unroll
    for (int ti = 0; ti < 2; ++ti)
#pragma unroll
      for (int tj = 0; tj < 4; ++tj)
        acc[ti][tj] = wmma_bf16(af[ti].v, bf[tj].v, acc[ti][tj]);
    __syncthreads();
  }

  // epilogue: section (q/k/v) is block-uniform -> branch-free scatter
  const int sec = N0 >> 10;
  u16* const dst = (sec == 0) ? Qb : ((sec == 1) ? Kb : Vb);
  const float scale = (sec == 0) ? 0.125f : 1.0f;   // fold 1/sqrt(64) into Q
#pragma unroll
  for (int ti = 0; ti < 2; ++ti) {
    const int mbase = M0 + wRow + ti * 16 + 8 * h;
#pragma unroll
    for (int tj = 0; tj < 4; ++tj) {
      const int n  = N0 + wCol + tj * 16 + l15;
      const float bv = bias[n];
      const int nc = n & 1023;
      const int head = nc >> 6, d = nc & 63;
#pragma unroll
      for (int i = 0; i < 8; ++i) {
        const int mr = mbase + i;
        const int bi = mr >> 11, s = mr & 2047;
        const float v = (acc[ti][tj][i] + bv) * scale;
        dst[((size_t)(bi * 16 + head) * 2048 + s) * 64 + d] = f2bf(v);
      }
    }
  }
}

// ---------------------------------------------------------------------------
// Kernel 2: flash attention per (b,h).  Double-buffered K (async-to-LDS) and
// V (register prefetch + transposed commit); one barrier per 32-key chunk.
// ---------------------------------------------------------------------------
__global__ __launch_bounds__(256) void attn_kernel(
    const u16* __restrict__ Qb, const u16* __restrict__ Kb,
    const u16* __restrict__ Vb, u16* __restrict__ Ob)
{
  __shared__ u16 Ks[2][32][72];   // [buf][key][hd] row-major
  __shared__ u16 Vt[2][64][40];   // [buf][hd][key] transposed
  __shared__ u16 Ps[8][16][40];   // per-wave P tile [q][k]

  const int tid = threadIdx.x;
  const int wid = tid >> 5, lane = tid & 31, l15 = lane & 15, h = lane >> 4;
  const int bh = blockIdx.y;
  const int b = bh >> 4, head = bh & 15;
  const int qBase = blockIdx.x * 128 + wid * 16;

  const u16* Qrow = Qb + ((size_t)bh * 2048 + qBase + l15) * 64;
  Frag16 qf[2];
  qf[0].q[0] = *(const u32x4*)(Qrow + 8 * h);
  qf[0].q[1] = *(const u32x4*)(Qrow + 16 + 8 * h);
  qf[1].q[0] = *(const u32x4*)(Qrow + 32 + 8 * h);
  qf[1].q[1] = *(const u32x4*)(Qrow + 48 + 8 * h);

  const v8f vzero = {0.f,0.f,0.f,0.f,0.f,0.f,0.f,0.f};
  v8f o[4]; o[0]=vzero; o[1]=vzero; o[2]=vzero; o[3]=vzero;
  float m[8], l[8];
#pragma unroll
  for (int i = 0; i < 8; ++i) { m[i] = -3.0e38f; l[i] = 0.f; }

  const u16* Kbase = Kb + (size_t)bh * 2048 * 64;
  const u16* Vbase = Vb + (size_t)bh * 2048 * 64;

  // per-thread staging coordinates
  const int skk = tid >> 3, sdd8 = (tid & 7) * 8;        // K: one b128 / thread
  int vk[2], vd[2];                                       // V: two u16x4 / thread
#pragma unroll
  for (int i = 0; i < 2; ++i) {
    int idx = tid + i * 256;
    vk[i] = idx >> 4; vd[i] = (idx & 15) * 4;
  }

  u16x4 vv[2];
  // ---- prologue: stage chunk 0 into buffer 0
#ifdef HAVE_ASYNC_LDS
  async_copy16(&Kbase[(size_t)skk * 64 + sdd8], &Ks[0][skk][sdd8]);
#else
  { u32x4 t = *(const u32x4*)&Kbase[(size_t)skk * 64 + sdd8];
    *(u32x4*)&Ks[0][skk][sdd8] = t; }
#endif
#pragma unroll
  for (int i = 0; i < 2; ++i)
    vv[i] = *(const u16x4*)&Vbase[(size_t)vk[i] * 64 + vd[i]];
#pragma unroll
  for (int i = 0; i < 2; ++i) {
    Vt[0][vd[i]+0][vk[i]] = vv[i][0]; Vt[0][vd[i]+1][vk[i]] = vv[i][1];
    Vt[0][vd[i]+2][vk[i]] = vv[i][2]; Vt[0][vd[i]+3][vk[i]] = vv[i][3];
  }
#ifdef HAVE_ASYNC_LDS
  asm volatile("s_wait_asynccnt 0" ::: "memory");
#endif
  __syncthreads();

  for (int it = 0; it < 64; ++it) {
    const int cur = it & 1, nxt = cur ^ 1;
    const bool more = (it + 1) < 64;
    const int kcn = (it + 1) * 32;

    // issue next chunk's K async copy + V register loads (overlap with compute)
    if (more) {
#ifdef HAVE_ASYNC_LDS
      async_copy16(&Kbase[(size_t)(kcn + skk) * 64 + sdd8], &Ks[nxt][skk][sdd8]);
#else
      u32x4 t = *(const u32x4*)&Kbase[(size_t)(kcn + skk) * 64 + sdd8];
      *(u32x4*)&Ks[nxt][skk][sdd8] = t;
#endif
#pragma unroll
      for (int i = 0; i < 2; ++i)
        vv[i] = *(const u16x4*)&Vbase[(size_t)(kcn + vk[i]) * 64 + vd[i]];
    }

    // scores: two 16q x 16k tiles, accumulate over hd=64 (two k=32 steps)
    v8f st[2];
#pragma unroll
    for (int t = 0; t < 2; ++t) {
      int key = t * 16 + l15;
      Frag16 kb0, kb1;
      kb0.q[0] = *(const u32x4*)&Ks[cur][key][8 * h];
      kb0.q[1] = *(const u32x4*)&Ks[cur][key][16 + 8 * h];
      kb1.q[0] = *(const u32x4*)&Ks[cur][key][32 + 8 * h];
      kb1.q[1] = *(const u32x4*)&Ks[cur][key][48 + 8 * h];
      v8f a = vzero;
      a = wmma_bf16(qf[0].v, kb0.v, a);
      a = wmma_bf16(qf[1].v, kb1.v, a);
      st[t] = a;
    }

    // online softmax
    float mx[8];
#pragma unroll
    for (int i = 0; i < 8; ++i) mx[i] = fmaxf(st[0][i], st[1][i]);
#pragma unroll
    for (int msk = 1; msk <= 8; msk <<= 1)
#pragma unroll
      for (int i = 0; i < 8; ++i) mx[i] = fmaxf(mx[i], __shfl_xor(mx[i], msk, 32));

    float rs[8];
#pragma unroll
    for (int i = 0; i < 8; ++i) {
      float mn = fmaxf(m[i], mx[i]);
      float al = __expf(m[i] - mn);
      m[i] = mn;
      float p0 = __expf(st[0][i] - mn);
      float p1 = __expf(st[1][i] - mn);
      st[0][i] = p0; st[1][i] = p1;
      rs[i] = p0 + p1;
      l[i] = l[i] * al;
      o[0][i] = o[0][i] * al; o[1][i] = o[1][i] * al;
      o[2][i] = o[2][i] * al; o[3][i] = o[3][i] * al;
    }
#pragma unroll
    for (int msk = 1; msk <= 8; msk <<= 1)
#pragma unroll
      for (int i = 0; i < 8; ++i) rs[i] += __shfl_xor(rs[i], msk, 32);
#pragma unroll
    for (int i = 0; i < 8; ++i) l[i] += rs[i];

    // C-layout -> A-layout transpose of P via per-wave LDS
#pragma unroll
    for (int t = 0; t < 2; ++t)
#pragma unroll
      for (int i = 0; i < 8; ++i)
        Ps[wid][i + 8 * h][t * 16 + l15] = f2bf(st[t][i]);
    asm volatile("s_wait_dscnt 0" ::: "memory");   // same-wave DS RAW fence

    Frag16 pf;
    pf.q[0] = *(const u32x4*)&Ps[wid][l15][8 * h];
    pf.q[1] = *(const u32x4*)&Ps[wid][l15][16 + 8 * h];
#pragma unroll
    for (int dt = 0; dt < 4; ++dt) {
      Frag16 vf;
      vf.q[0] = *(const u32x4*)&Vt[cur][dt * 16 + l15][8 * h];
      vf.q[1] = *(const u32x4*)&Vt[cur][dt * 16 + l15][16 + 8 * h];
      o[dt] = wmma_bf16(pf.v, vf.v, o[dt]);
    }

    // commit next V chunk (transposed) into the alternate buffer
    if (more) {
#pragma unroll
      for (int i = 0; i < 2; ++i) {
        Vt[nxt][vd[i]+0][vk[i]] = vv[i][0]; Vt[nxt][vd[i]+1][vk[i]] = vv[i][1];
        Vt[nxt][vd[i]+2][vk[i]] = vv[i][2]; Vt[nxt][vd[i]+3][vk[i]] = vv[i][3];
      }
#ifdef HAVE_ASYNC_LDS
      asm volatile("s_wait_asynccnt 0" ::: "memory");
#endif
    }
    __syncthreads();
  }

  // normalize and store bf16 [B][S][D]
  u16* const obase = Ob + (size_t)b * 2048 * 1024 + head * 64;
#pragma unroll
  for (int i = 0; i < 8; ++i) {
    float inv = 1.0f / l[i];
    int s = qBase + i + 8 * h;
#pragma unroll
    for (int dt = 0; dt < 4; ++dt)
      obase[(size_t)s * 1024 + dt * 16 + l15] = f2bf(o[dt][i] * inv);
  }
}

// ---------------------------------------------------------------------------
// Kernel 3: out = attn @ W_out + b_out  (A already bf16, output f32)
// ---------------------------------------------------------------------------
__global__ __launch_bounds__(256) void out_gemm_kernel(
    const u16* __restrict__ Ab, const float* __restrict__ W,
    const float* __restrict__ bias, float* __restrict__ out)
{
  __shared__ u16 As[128][40];
  __shared__ u16 Bt[128][40];

  const int tid = threadIdx.x;
  const int wid = tid >> 5, lane = tid & 31, l15 = lane & 15, h = lane >> 4;
  const int wRow = (wid >> 1) * 32, wCol = (wid & 1) * 64;
  const int M0 = blockIdx.x * 128, N0 = blockIdx.y * 128;

  const v8f vzero = {0.f,0.f,0.f,0.f,0.f,0.f,0.f,0.f};
  v8f acc[2][4];
#pragma unroll
  for (int ti = 0; ti < 2; ++ti)
#pragma unroll
    for (int tj = 0; tj < 4; ++tj) acc[ti][tj] = vzero;

  u16x4 aPre[4]; f32x4 bPre[4];
#pragma unroll
  for (int i = 0; i < 4; ++i) {
    int idx = tid + i * 256;
    int row = idx >> 3, kk4 = (idx & 7) * 4;
    aPre[i] = *(const u16x4*)&Ab[(size_t)(M0 + row) * 1024 + kk4];
    int kk = idx >> 5, nn4 = (idx & 31) * 4;
    bPre[i] = *(const f32x4*)&W[(size_t)kk * 1024 + N0 + nn4];
  }

  for (int k0 = 0; k0 < 1024; k0 += 32) {
#pragma unroll
    for (int i = 0; i < 4; ++i) {
      int idx = tid + i * 256;
      int row = idx >> 3, kk4 = (idx & 7) * 4;
      *(u16x4*)&As[row][kk4] = aPre[i];
      int kk = idx >> 5, nn4 = (idx & 31) * 4;
      Bt[nn4+0][kk]=f2bf(bPre[i][0]); Bt[nn4+1][kk]=f2bf(bPre[i][1]);
      Bt[nn4+2][kk]=f2bf(bPre[i][2]); Bt[nn4+3][kk]=f2bf(bPre[i][3]);
    }
    __syncthreads();

    if (k0 + 32 < 1024) {
      int kn = k0 + 32;
#pragma unroll
      for (int i = 0; i < 4; ++i) {
        int idx = tid + i * 256;
        int row = idx >> 3, kk4 = (idx & 7) * 4;
        aPre[i] = *(const u16x4*)&Ab[(size_t)(M0 + row) * 1024 + kn + kk4];
        int kk = idx >> 5, nn4 = (idx & 31) * 4;
        bPre[i] = *(const f32x4*)&W[(size_t)(kn + kk) * 1024 + N0 + nn4];
      }
    }

    Frag16 af[2], bf[4];
#pragma unroll
    for (int t = 0; t < 2; ++t) {
      int row = wRow + t * 16 + l15;
      af[t].q[0] = *(const u32x4*)&As[row][8 * h];
      af[t].q[1] = *(const u32x4*)&As[row][16 + 8 * h];
    }
#pragma unroll
    for (int t = 0; t < 4; ++t) {
      int col = wCol + t * 16 + l15;
      bf[t].q[0] = *(const u32x4*)&Bt[col][8 * h];
      bf[t].q[1] = *(const u32x4*)&Bt[col][16 + 8 * h];
    }
#pragma unroll
    for (int ti = 0; ti < 2; ++ti)
#pragma unroll
      for (int tj = 0; tj < 4; ++tj)
        acc[ti][tj] = wmma_bf16(af[ti].v, bf[tj].v, acc[ti][tj]);
    __syncthreads();
  }

#pragma unroll
  for (int ti = 0; ti < 2; ++ti) {
    const int mbase = M0 + wRow + ti * 16 + 8 * h;
#pragma unroll
    for (int tj = 0; tj < 4; ++tj) {
      const int n = N0 + wCol + tj * 16 + l15;
      const float bv = bias[n];
#pragma unroll
      for (int i = 0; i < 8; ++i)
        out[(size_t)(mbase + i) * 1024 + n] = acc[ti][tj][i] + bv;
    }
  }
}

// ---------------------------------------------------------------------------
extern "C" void kernel_launch(void* const* d_in, const int* in_sizes, int n_in,
                              void* d_out, int out_size, void* d_ws, size_t ws_size,
                              hipStream_t stream)
{
  const float* x    = (const float*)d_in[0];
  const float* Wqkv = (const float*)d_in[1];
  const float* bqkv = (const float*)d_in[2];
  const float* Wout = (const float*)d_in[3];
  const float* bout = (const float*)d_in[4];
  float* out = (float*)d_out;

  const size_t NE = (size_t)4 * 16 * 2048 * 64;   // 8,388,608 elems per tensor
  u16* Qb = (u16*)d_ws;        // bf16 Q, pre-scaled by 1/sqrt(hd)
  u16* Kb = Qb + NE;
  u16* Vb = Kb + NE;
  u16* Ab = Vb + NE;           // bf16 attention output [B][S][D]

  qkv_gemm_kernel<<<dim3(64, 24), 256, 0, stream>>>(x, Wqkv, bqkv, Qb, Kb, Vb);
  attn_kernel    <<<dim3(16, 64), 256, 0, stream>>>(Qb, Kb, Vb, Ab);
  out_gemm_kernel<<<dim3(64,  8), 256, 0, stream>>>(Ab, Wout, bout, out);

  (void)in_sizes; (void)n_in; (void)out_size; (void)ws_size;
}
